// PhaseDifferenceEdgeComputer_558345748624
// MI455X (gfx1250) — matrix operand order repaired and checked by reading, not compiled, not run on hardware
//
#include <hip/hip_runtime.h>

// ---------------------------------------------------------------------------
// PhaseDifferenceEdgeComputer for MI455X (gfx1250, wave32, WMMA)
// B=2048, S=8, N=32, D=64. Pairs per batch = 496 = 31 tiles of 16 rows.
// Layer1: x(16x160 bf16, K padded 134->160) @ W1(160x64)  -> 20 WMMAs/wave
// Layer2: h(16x64 bf16) @ W2(64x32)                        ->  4 WMMAs/wave
// Layer3: 32-wide dot + sigmoid on VALU.
// ---------------------------------------------------------------------------

typedef float  v4f   __attribute__((ext_vector_type(4)));
typedef float  v8f   __attribute__((ext_vector_type(8)));
typedef __bf16 v16bf __attribute__((ext_vector_type(16)));

union FragU { v4f f4[2]; v16bf bf; };

#define NNODE 32
#define DFEAT 64
#define KPAD 160   // 134 padded to 5 * 32

__device__ __forceinline__ unsigned short f2bf(float f) {
    union { float f; unsigned u; } x; x.f = f;
    unsigned r = x.u + 0x7FFFu + ((x.u >> 16) & 1u);   // round-to-nearest-even
    return (unsigned short)(r >> 16);
}

__device__ __forceinline__ float gelu_exact(float x) {
    return 0.5f * x * (1.0f + erff(x * 0.70710678118654752f));
}

__device__ __forceinline__ float sigmoidf_(float x) {
    return 1.0f / (1.0f + expf(-x));
}

// ---------------------------------------------------------------------------
// Prep: repack W1 (134x64, zero-padded to 160x64) and W2 (64x32) as bf16
// B-matrix WMMA fragments. Fragment f: 32 lanes x 8 dwords (1 KB), lane holds
// column n = nc*16 + (lane&15); VGPR v holds K = kc*32 + 16*(lane>>4) + 2v,+1.
// Layout in ws: frags 0..19 = W1 (kc*4+nc), frags 20..23 = W2 (20 + kc*2+nc).
// ---------------------------------------------------------------------------
__global__ void prep_weight_frags(const float* __restrict__ W1,
                                  const float* __restrict__ W2,
                                  unsigned* __restrict__ ws) {
    int e = blockIdx.x * blockDim.x + threadIdx.x;
    if (e >= 24 * 256) return;
    int f = e >> 8, rem = e & 255, lane = rem >> 3, v = rem & 7;
    int half = lane >> 4, col = lane & 15;
    unsigned w;
    if (f < 20) {
        int kc = f >> 2, nc = f & 3;
        int k = kc * 32 + half * 16 + 2 * v;
        int n = nc * 16 + col;
        float a = (k     < 134) ? W1[k * 64 + n]       : 0.0f;
        float b = (k + 1 < 134) ? W1[(k + 1) * 64 + n] : 0.0f;
        w = (unsigned)f2bf(a) | ((unsigned)f2bf(b) << 16);
    } else {
        int g = f - 20, kc = g >> 1, nc = g & 1;
        int k = kc * 32 + half * 16 + 2 * v;
        int n = nc * 16 + col;
        float a = W2[k * 32 + n];
        float b = W2[(k + 1) * 32 + n];
        w = (unsigned)f2bf(a) | ((unsigned)f2bf(b) << 16);
    }
    ws[f * 256 + lane * 8 + v] = w;
}

// Per-wave LDS scratch. H/H2 overlay A (union => compiler keeps DS order; the
// LDS pipe is in-order per wave anyway).
struct WaveScratch {
    union {
        unsigned short A[16][KPAD];                     // 5120 B (layer-1 x)
        struct {
            unsigned short H[16][DFEAT];                // 2048 B (layer-1 out, bf16)
            float          H2[16][32];                  // 2048 B (layer-2 out, f32)
        } s;
    };
};

__global__ __launch_bounds__(256)
void edge_mlp_kernel(const float* __restrict__ cel,  const float* __restrict__ theta,
                     const float* __restrict__ phi,  const float* __restrict__ vel,
                     const float* __restrict__ rad,  const float* __restrict__ lon,
                     const float* __restrict__ b1,   const float* __restrict__ b2,
                     const float* __restrict__ W3,   const float* __restrict__ b3,
                     const float* __restrict__ pw,   const unsigned* __restrict__ wfrag,
                     float* __restrict__ out, int S) {
    __shared__ __align__(16) unsigned short featbf[NNODE][DFEAT];   // 4 KB
    __shared__ float sth[NNODE], sph[NNODE], svl[NNODE], srd[NNODE], slg[NNODE];
    __shared__ WaveScratch wsc[8];                                   // 40 KB

    const int tid = threadIdx.x;
    const int b   = blockIdx.x >> 2;     // batch
    const int q   = blockIdx.x & 3;      // quarter of the 31 pair-tiles

    // ---- phase 0: stage feats (last S slice) as bf16 + per-node scalars ----
    {
        const float* src = cel + ((size_t)b * S + (S - 1)) * (NNODE * DFEAT);
        #pragma unroll
        for (int r = 0; r < 8; ++r) {
            int idx = tid + r * 256;                       // 0..2047
            featbf[idx >> 6][idx & 63] = f2bf(src[idx]);
        }
        if (tid < NNODE * 5) {
            int arr = tid >> 5, l = tid & 31;
            const float* sa = arr == 0 ? theta : arr == 1 ? phi : arr == 2 ? vel
                                                : arr == 3 ? rad : lon;
            float* da = arr == 0 ? sth : arr == 1 ? sph : arr == 2 ? svl
                                        : arr == 3 ? srd : slg;
            da[l] = sa[b * NNODE + l];
        }
        if (q == 0 && tid < NNODE) out[(size_t)b * 1024 + tid * 33] = 0.0f; // diag
    }
    __syncthreads();

    const int w    = tid >> 5;           // wave in block
    const int lane = tid & 31;
    const int t    = q * 8 + w;          // pair-tile index
    if (t >= 31) return;                 // uniform per wave

    const int m    = lane & 15;          // A-row / C-column index for this lane
    const int half = lane >> 4;
    const int klo  = half * 16;

    // pair (i,j) for row m of this tile
    int p = t * 16 + m;
    int i = 0, rem = p;
    while (rem >= (NNODE - 1 - i)) { rem -= (NNODE - 1 - i); ++i; }
    int j = i + 1 + rem;

    // ---- build A tile: row m = [ feats[i] | feats[j] | 6 scalars | 0-pad ] ----
    {
        unsigned short* dst = &wsc[w].A[m][0];
        const v4f* s0 = (const v4f*)&featbf[i][half * 32];
        const v4f* s1 = (const v4f*)&featbf[j][half * 32];
        v4f* d0 = (v4f*)(dst + half * 32);
        v4f* d1 = (v4f*)(dst + 64 + half * 32);
        #pragma unroll
        for (int c = 0; c < 4; ++c) { d0[c] = s0[c]; d1[c] = s1[c]; }
    }

    float gate = 0.0f;
    if (half == 0) {
        float dth = sth[i] - sth[j];  float td = atan2f(sinf(dth), cosf(dth));
        float dph = sph[i] - sph[j];  float pd = atan2f(sinf(dph), cosf(dph));
        float vd  = svl[i] - svl[j];
        float rr  = srd[i] / (srd[j] + 1e-8f);
        float dlg = slg[i] - slg[j];  float ld = atan2f(sinf(dlg), cosf(dlg));
        float pr  = cosf(td) * cosf(pd);
        unsigned short* s = &wsc[w].A[m][128];
        s[0] = f2bf(td); s[1] = f2bf(pd); s[2] = f2bf(vd);
        s[3] = f2bf(rr); s[4] = f2bf(ld); s[5] = f2bf(pr);
        unsigned* z = (unsigned*)(s + 6);                 // cols 134..159 -> 0
        #pragma unroll
        for (int c = 0; c < 13; ++c) z[c] = 0u;
        float g = fabsf(td) * pw[0] + fabsf(pd) * pw[1] + fabsf(vd) * pw[2]
                + fabsf(rr - 1.0f) * pw[3] + fabsf(ld) * pw[4] + fabsf(pr) * pw[5];
        gate = sigmoidf_(g);
    }

    // ---- layer 1: 5 K-chunks x 4 N-chunks of v_wmma_f32_16x16x32_bf16 ----
    v8f acc0 = {}, acc1 = {}, acc2 = {}, acc3 = {};
    #pragma unroll
    for (int kc = 0; kc < 5; ++kc) {
        FragU a;
        const unsigned short* ar = &wsc[w].A[m][kc * 32];
        a.f4[0] = *(const v4f*)(ar + klo);        // VGPR 0-3: K = kc*32+klo..+7
        a.f4[1] = *(const v4f*)(ar + klo + 8);    // VGPR 4-7: K = +8
        #pragma unroll
        for (int nc = 0; nc < 4; ++nc) {
            FragU bf;
            const v4f* bp = (const v4f*)(wfrag + ((kc * 4 + nc) * 256 + lane * 8));
            bf.f4[0] = bp[0]; bf.f4[1] = bp[1];
            v8f& acc = nc == 0 ? acc0 : nc == 1 ? acc1 : nc == 2 ? acc2 : acc3;
            acc = __builtin_amdgcn_wmma_f32_16x16x32_bf16(
                false, a.bf, false, bf.bf, (short)0, acc, false, false);
        }
    }

    // ---- bias + exact GELU, store h as bf16 in WMMA-A-readable layout ----
    #pragma unroll
    for (int nc = 0; nc < 4; ++nc) {
        v8f acc = nc == 0 ? acc0 : nc == 1 ? acc1 : nc == 2 ? acc2 : acc3;
        int n = nc * 16 + m;                      // C: column = lane&15
        float bias = b1[n];
        #pragma unroll
        for (int r = 0; r < 8; ++r)               // C: row = r + 8*half
            wsc[w].s.H[r + half * 8][n] = f2bf(gelu_exact(acc[r] + bias));
    }

    // ---- layer 2: 2 K-chunks x 2 N-chunks ----
    v8f c0 = {}, c1 = {};
    #pragma unroll
    for (int kc = 0; kc < 2; ++kc) {
        FragU a;
        const unsigned short* ar = &wsc[w].s.H[m][kc * 32];
        a.f4[0] = *(const v4f*)(ar + klo);
        a.f4[1] = *(const v4f*)(ar + klo + 8);
        #pragma unroll
        for (int nc = 0; nc < 2; ++nc) {
            FragU bf;
            const v4f* bp = (const v4f*)(wfrag + ((20 + kc * 2 + nc) * 256 + lane * 8));
            bf.f4[0] = bp[0]; bf.f4[1] = bp[1];
            v8f& acc = nc == 0 ? c0 : c1;
            acc = __builtin_amdgcn_wmma_f32_16x16x32_bf16(
                false, a.bf, false, bf.bf, (short)0, acc, false, false);
        }
    }
    #pragma unroll
    for (int nc = 0; nc < 2; ++nc) {
        v8f acc = nc == 0 ? c0 : c1;
        int n = nc * 16 + m;
        float bias = b2[n];
        #pragma unroll
        for (int r = 0; r < 8; ++r)
            wsc[w].s.H2[r + half * 8][n] = gelu_exact(acc[r] + bias);
    }

    // ---- layer 3: 32-wide dot + sigmoid, gated symmetric scatter ----
    if (half == 0) {
        float s = b3[0];
        #pragma unroll
        for (int n = 0; n < 32; ++n) s += wsc[w].s.H2[m][n] * W3[n];
        float fe = sigmoidf_(s) * gate;
        size_t base = (size_t)b * 1024;
        out[base + i * 32 + j] = fe;
        out[base + j * 32 + i] = fe;
    }
}

extern "C" void kernel_launch(void* const* d_in, const int* in_sizes, int n_in,
                              void* d_out, int out_size, void* d_ws, size_t ws_size,
                              hipStream_t stream) {
    const float* cel   = (const float*)d_in[0];
    const float* theta = (const float*)d_in[1];
    const float* phi   = (const float*)d_in[2];
    const float* vel   = (const float*)d_in[3];
    const float* rad   = (const float*)d_in[4];
    const float* lon   = (const float*)d_in[5];
    const float* W1    = (const float*)d_in[6];
    const float* b1    = (const float*)d_in[7];
    const float* W2    = (const float*)d_in[8];
    const float* b2    = (const float*)d_in[9];
    const float* W3    = (const float*)d_in[10];
    const float* b3    = (const float*)d_in[11];
    const float* pw    = (const float*)d_in[12];
    float* out = (float*)d_out;

    if (ws_size < (size_t)(24 * 1024)) return;   // need 24 KB for weight frags
    unsigned* wfrag = (unsigned*)d_ws;

    int B = in_sizes[1] / NNODE;                             // 2048
    int S = in_sizes[0] / (B * NNODE * DFEAT);               // 8

    prep_weight_frags<<<24, 256, 0, stream>>>(W1, W2, wfrag);
    edge_mlp_kernel<<<dim3((unsigned)B * 4), dim3(256), 0, stream>>>(
        cel, theta, phi, vel, rad, lon, b1, b2, W3, b3, pw, wfrag, out, S);
}
